// QuantFA_27513560499034
// MI455X (gfx1250) — 8-bit matrix operand order repaired and checked by reading, compiled, not run on hardware
//
#include <hip/hip_runtime.h>
#include <hip/hip_bf16.h>

#define DH   128
#define SQ   1024
#define NH   24
#define NB   8
#define BQ   128
#define BKV  64

#define QS_OFF  0
#define KS_OFF  16384
#define VS_OFF  24576
#define PS_OFF  32768
#define SMEM_BYTES 49152

typedef __attribute__((ext_vector_type(8))) int      v8i_t;
typedef __attribute__((ext_vector_type(8))) float    v8f_t;
typedef __attribute__((ext_vector_type(4))) unsigned u32x4_t;
typedef __attribute__((ext_vector_type(8))) int      i32x8_t;
typedef __attribute__((ext_vector_type(4))) int      i32x4_t;

#if __has_builtin(__builtin_amdgcn_tensor_load_to_lds)
#define HAVE_TDM 1
__device__ __forceinline__ void tdm_load_tile(unsigned lds_off, const void* gptr,
                                              unsigned tile_d0_u8,
                                              unsigned tile_d1,
                                              unsigned stride_u8) {
  const unsigned long long ga = (unsigned long long)(uintptr_t)gptr;
  u32x4_t g0;
  g0[0] = 1u;
  g0[1] = lds_off;
  g0[2] = (unsigned)ga;
  g0[3] = (unsigned)((ga >> 32) & 0x01ffffffu)
          | (2u << 30);
  const unsigned dim0 = 1u << 20;
  const unsigned dim1 = 1u << 20;
  i32x8_t g1;
  g1[0] = (int)(3u << 16);
  g1[1] = (int)((dim0 & 0xffffu) << 16);
  g1[2] = (int)(((dim0 >> 16) & 0xffffu) | ((dim1 & 0xffffu) << 16));
  g1[3] = (int)(((dim1 >> 16) & 0xffffu) | ((tile_d0_u8 & 0xffffu) << 16));
  g1[4] = (int)(tile_d1 & 0xffffu);
  g1[5] = (int)stride_u8;
  g1[6] = 0;
  g1[7] = 0;
  const i32x4_t zz = {0, 0, 0, 0};
#if __clang_major__ >= 23
  const i32x8_t z8 = {0, 0, 0, 0, 0, 0, 0, 0};
  __builtin_amdgcn_tensor_load_to_lds(g0, g1, zz, zz, z8, 0);
#else
  __builtin_amdgcn_tensor_load_to_lds(g0, g1, zz, zz, 0);
#endif
}
#endif

__device__ __forceinline__ int quant_i8(float x, float inv) {
  float r = rintf(x * inv);
  r = fminf(fmaxf(r, -128.f), 127.f);
  return ((int)r) & 0xff;
}

__device__ __forceinline__ v8i_t loadA8(const char* base, int stride, int row,
                                        int half, int koff) {
  v8i_t a;
  const char* p = base + row * stride + koff + half * 16;
#pragma unroll
  for (int j = 0; j < 4; ++j) {
    unsigned long long u = *(const unsigned long long*)(p + (j >> 1) * 32 + (j & 1) * 8);
    a[2 * j]     = (int)(unsigned)(u & 0xffffffffull);
    a[2 * j + 1] = (int)(unsigned)(u >> 32);
  }
  return a;
}

__device__ __forceinline__ v8i_t loadB8(const char* base, int stride, int col,
                                        int half, int koff) {
  const char* p = base + col * stride + koff + half * 16;
  const int4 x = *(const int4*)(p);
  const int4 y = *(const int4*)(p + 32);
  v8i_t b;
  b[0] = x.x; b[1] = x.y; b[2] = x.z; b[3] = x.w;
  b[4] = y.x; b[5] = y.y; b[6] = y.z; b[7] = y.w;
  return b;
}

__global__ __launch_bounds__(256) void quant_rm_kernel(
    const float* __restrict__ src, const float* __restrict__ scale,
    unsigned* __restrict__ dst) {
  int idx = blockIdx.x * 256 + threadIdx.x;
  int d4  = idx & 31;
  int n   = (idx >> 5) % NH;
  int t   = idx / (32 * NH);
  int b   = t / SQ, s = t % SQ;
  float inv = 1.0f / scale[n];
  const float4 x = ((const float4*)src)[idx];
  unsigned r = (unsigned)quant_i8(x.x, inv) |
               ((unsigned)quant_i8(x.y, inv) << 8) |
               ((unsigned)quant_i8(x.z, inv) << 16) |
               ((unsigned)quant_i8(x.w, inv) << 24);
  dst[(((size_t)(b * NH + n) * SQ + s) << 5) + d4] = r;
}

__global__ __launch_bounds__(256) void quant_vt_kernel(
    const float* __restrict__ v, const float* __restrict__ scale,
    char* __restrict__ vt) {
  __shared__ char Tt[128 * 72];
  const int bn = blockIdx.y, st = blockIdx.x;
  const int n = bn % NH, b = bn / NH;
  const float inv = 1.0f / scale[n];
#pragma unroll
  for (int i = 0; i < 8; ++i) {
    int e4   = threadIdx.x + i * 256;
    int d4   = e4 & 31;
    int srow = e4 >> 5;
    const float4 x = *(const float4*)(
        v + (((size_t)(b * SQ + st * 64 + srow) * NH + n) * DH + d4 * 4));
    Tt[(d4 * 4 + 0) * 72 + srow] = (char)quant_i8(x.x, inv);
    Tt[(d4 * 4 + 1) * 72 + srow] = (char)quant_i8(x.y, inv);
    Tt[(d4 * 4 + 2) * 72 + srow] = (char)quant_i8(x.z, inv);
    Tt[(d4 * 4 + 3) * 72 + srow] = (char)quant_i8(x.w, inv);
  }
  __syncthreads();
  const int d = threadIdx.x >> 1, c = threadIdx.x & 1;
  unsigned long long* dst = (unsigned long long*)(
      vt + (size_t)bn * DH * SQ + (size_t)d * SQ + st * 64 + c * 32);
  const unsigned long long* srcp =
      (const unsigned long long*)(Tt + d * 72 + c * 32);
#pragma unroll
  for (int i = 0; i < 4; ++i) dst[i] = srcp[i];
}

__global__ __launch_bounds__(256) void attn_i8_kernel(
    const char* __restrict__ q8p, const char* __restrict__ k8p,
    const char* __restrict__ vtp, const float* __restrict__ qsc,
    const float* __restrict__ ksc, const float* __restrict__ vsc,
    float* __restrict__ out) {
  __shared__ __align__(16) char smem[SMEM_BYTES];
  char* Qs = smem + QS_OFF;
  char* Ks = smem + KS_OFF;
  char* Vs = smem + VS_OFF;
  char* Ps = smem + PS_OFF;

  const int bn = blockIdx.y, qt = blockIdx.x;
  const int n = bn % NH, b = bn / NH;
  const int tid = threadIdx.x, wave = tid >> 5, lane = tid & 31;
  const int lanelo = lane & 15, half = lane >> 4;

  const float sQK = qsc[n] * ksc[n] * 0.08838834764831845f;
  const float sV  = vsc[n] * (1.0f / 127.0f);

  const char* qg = q8p + ((size_t)bn * SQ + (size_t)qt * BQ) * DH;
  const char* kg = k8p + (size_t)bn * SQ * DH;
  const char* vg = vtp + (size_t)bn * DH * SQ;

#if HAVE_TDM
  if (wave == 0) {
    tdm_load_tile(QS_OFF, qg, (BQ * DH) / 8, 0, (BQ * DH) / 8);
    __builtin_amdgcn_s_wait_tensorcnt(0);
  }
#else
#pragma unroll
  for (int i = 0; i < 4; ++i)
    ((int4*)Qs)[tid + i * 256] = ((const int4*)qg)[tid + i * 256];
#endif
  __syncthreads();

  const v8i_t aq0 = loadA8(Qs, DH, wave * 16 + lanelo, half, 0);
  const v8i_t aq1 = loadA8(Qs, DH, wave * 16 + lanelo, half, 64);

  v8f_t o[8];
  float mrun[8], lrun[8];
#pragma unroll
  for (int e = 0; e < 8; ++e) {
    mrun[e] = -1e30f;
    lrun[e] = 0.f;
#pragma unroll
    for (int dt = 0; dt < 8; ++dt) o[dt][e] = 0.f;
  }

  char* Psw = Ps + wave * (16 * BKV);

  for (int kt = 0; kt < SQ / BKV; ++kt) {
    __syncthreads();
#if HAVE_TDM
    if (wave == 0) {
      tdm_load_tile(KS_OFF, kg + (size_t)kt * BKV * DH, (BKV * DH) / 8, 0,
                    (BKV * DH) / 8);
      tdm_load_tile(VS_OFF, vg + kt * BKV, BKV / 8, DH, SQ / 8);
      __builtin_amdgcn_s_wait_tensorcnt(0);
    }
    if (kt + 1 < SQ / BKV)
      __builtin_prefetch(kg + (size_t)(kt + 1) * BKV * DH + tid * 32, 0, 1);
#else
    ((int4*)Ks)[tid * 2]     = ((const int4*)(kg + (size_t)kt * BKV * DH))[tid * 2];
    ((int4*)Ks)[tid * 2 + 1] = ((const int4*)(kg + (size_t)kt * BKV * DH))[tid * 2 + 1];
    {
      const int d = tid >> 1, c = tid & 1;
      const int4* vrow = (const int4*)(vg + (size_t)d * SQ + kt * BKV + c * 32);
      int4* vdst = (int4*)(Vs + d * BKV + c * 32);
      vdst[0] = vrow[0];
      vdst[1] = vrow[1];
    }
#endif
    __syncthreads();

    v8i_t bf0[4], bf1[4];
#pragma unroll
    for (int ct = 0; ct < 4; ++ct) {
      bf0[ct] = loadB8(Ks, DH, ct * 16 + lanelo, half, 0);
      bf1[ct] = loadB8(Ks, DH, ct * 16 + lanelo, half, 64);
    }
    v8i_t acc[4];
#pragma unroll
    for (int ct = 0; ct < 4; ++ct) {
      v8i_t zz;
#pragma unroll
      for (int e = 0; e < 8; ++e) zz[e] = 0;
      acc[ct] = __builtin_amdgcn_wmma_i32_16x16x64_iu8(true, aq0, true, bf0[ct],
                                                       zz, false, false);
    }
#pragma unroll
    for (int ct = 0; ct < 4; ++ct)
      acc[ct] = __builtin_amdgcn_wmma_i32_16x16x64_iu8(true, aq1, true, bf1[ct],
                                                       acc[ct], false, false);
    v8f_t sf[4];
#pragma unroll
    for (int ct = 0; ct < 4; ++ct)
#pragma unroll
      for (int e = 0; e < 8; ++e) sf[ct][e] = (float)acc[ct][e] * sQK;

    float alpha[8];
#pragma unroll
    for (int e = 0; e < 8; ++e) {
      float mv = fmaxf(fmaxf(sf[0][e], sf[1][e]), fmaxf(sf[2][e], sf[3][e]));
#pragma unroll
      for (int mk = 1; mk < 16; mk <<= 1) mv = fmaxf(mv, __shfl_xor(mv, mk, 32));
      const float mnew = fmaxf(mrun[e], mv);
      alpha[e] = __expf(mrun[e] - mnew);
      float rs = 0.f;
#pragma unroll
      for (int ct = 0; ct < 4; ++ct) {
        const float p = __expf(sf[ct][e] - mnew);
        sf[ct][e] = p;
        rs += p;
      }
#pragma unroll
      for (int mk = 1; mk < 16; mk <<= 1) rs += __shfl_xor(rs, mk, 32);
      lrun[e] = lrun[e] * alpha[e] + rs;
      mrun[e] = mnew;
    }

#pragma unroll
    for (int ct = 0; ct < 4; ++ct)
#pragma unroll
      for (int e = 0; e < 8; ++e) {
        const int pq = (int)(sf[ct][e] * 127.f + 0.5f);
        Psw[(e + 8 * half) * BKV + ct * 16 + lanelo] = (char)pq;
      }
#pragma unroll
    for (int dt = 0; dt < 8; ++dt)
#pragma unroll
      for (int e = 0; e < 8; ++e) o[dt][e] *= alpha[e];

    const v8i_t ap = loadA8(Psw, BKV, lanelo, half, 0);
#pragma unroll
    for (int dt = 0; dt < 8; ++dt) {
      const v8i_t bv = loadB8(Vs, BKV, dt * 16 + lanelo, half, 0);
      v8i_t z;
#pragma unroll
      for (int e = 0; e < 8; ++e) z[e] = 0;
      const v8i_t t = __builtin_amdgcn_wmma_i32_16x16x64_iu8(false, ap, true,
                                                             bv, z, false, false);
#pragma unroll
      for (int e = 0; e < 8; ++e) o[dt][e] += (float)t[e];
    }
  }

#pragma unroll
  for (int e = 0; e < 8; ++e) {
    const int m = e + 8 * half;
    const size_t tg = (size_t)b * SQ + qt * BQ + wave * 16 + m;
    const float f = sV / lrun[e];
#pragma unroll
    for (int dt = 0; dt < 8; ++dt)
      out[(tg * NH + n) * DH + dt * 16 + lanelo] = o[dt][e] * f;
  }
}

extern "C" void kernel_launch(void* const* d_in, const int* in_sizes, int n_in,
                              void* d_out, int out_size, void* d_ws,
                              size_t ws_size, hipStream_t stream) {
  (void)in_sizes; (void)n_in; (void)out_size; (void)ws_size;
  const float* q   = (const float*)d_in[0];
  const float* k   = (const float*)d_in[1];
  const float* v   = (const float*)d_in[2];
  const float* qsc = (const float*)d_in[3];
  const float* ksc = (const float*)d_in[4];
  const float* vsc = (const float*)d_in[5];

  const size_t QB = (size_t)NB * SQ * NH * DH;
  char* q8 = (char*)d_ws;
  char* k8 = q8 + QB;
  char* vt = k8 + QB;

  const int quant_blocks = (int)(QB / 4 / 256);
  quant_rm_kernel<<<quant_blocks, 256, 0, stream>>>(q, qsc, (unsigned*)q8);
  quant_rm_kernel<<<quant_blocks, 256, 0, stream>>>(k, ksc, (unsigned*)k8);
  quant_vt_kernel<<<dim3(SQ / 64, NB * NH), 256, 0, stream>>>(v, vsc, vt);

  attn_i8_kernel<<<dim3(SQ / BQ, NB * NH), 256, 0, stream>>>(
      q8, k8, vt, qsc, ksc, vsc, (float*)d_out);
}
